// QuantLinear_2920577761341
// MI455X (gfx1250) — matrix operand order repaired and checked by reading, not compiled, not run on hardware
//
#include <hip/hip_runtime.h>
#include <hip/hip_bf16.h>

// ---------------------------------------------------------------------------
// QuantLinear for MI455X (gfx1250, wave32, WMMA):
//   Pass 1: dequantize 4-bit weights (+group scales, +cherry outliers) into a
//           dense f16 weight matrix in d_ws (O x I, 33.5 MB).
//   Pass 2: LDS-tiled f16 WMMA GEMM (f32 accumulate), 128x128x32 block tiles,
//           8 wave32 per block, v_wmma_f32_16x16x32_f16, double-buffered LDS.
//           B tiles (already f16) go global->LDS via GLOBAL_LOAD_ASYNC_TO_LDS
//           (ASYNCcnt), A tiles (f32 x) are staged through VGPRs + cvt to f16.
// ---------------------------------------------------------------------------

typedef __attribute__((ext_vector_type(16))) _Float16 v16h;
typedef __attribute__((ext_vector_type(8)))  _Float16 v8h;
typedef __attribute__((ext_vector_type(8)))  float    v8f;

#define BM 128
#define BN 128
#define BK 32
#define LDSS 40   // padded LDS row stride in f16 elems (32 + 8) => 80B, conflict-free b128 reads

// ---------------------------------------------------------------------------
// Pass 1: dequantize one output row per block.
// ---------------------------------------------------------------------------
__global__ __launch_bounds__(128) void dequant_rows_kernel(
    const int* __restrict__ qweight,      // (O, packed) one byte per int32
    const float* __restrict__ scales,     // (O, n_groups)
    const int* __restrict__ cidx,         // (O, nch)
    const float* __restrict__ cw,         // (O, nch)
    _Float16* __restrict__ Wh,            // (O, I) output
    int O, int I, int packed, int n_groups, int nch)
{
    __shared__ int   s_ci[128];
    __shared__ float s_cw[128];
    const int o = blockIdx.x;
    for (int j = threadIdx.x; j < nch; j += blockDim.x) {
        s_ci[j] = cidx[(size_t)o * nch + j];
        s_cw[j] = cw[(size_t)o * nch + j];
    }
    __syncthreads();

    const int*   qrow = qweight + (size_t)o * packed;
    const float* srow = scales  + (size_t)o * n_groups;
    _Float16*    wrow = Wh      + (size_t)o * I;

    for (int c = threadIdx.x; c < I; c += blockDim.x) {
        int rank = 0, slot = -1;
        for (int j = 0; j < nch; ++j) {
            const int ci = s_ci[j];
            rank += (ci < c) ? 1 : 0;
            if (ci == c) slot = j;
        }
        float val;
        if (slot >= 0) {
            val = s_cw[slot];
        } else {
            const int jn   = c - rank;                    // normal index
            const int byte = qrow[jn >> 1] & 0xFF;        // two nibbles per byte
            const int nib  = (byte >> (4 * (jn & 1))) & 0xF;
            val = (float)(nib - 8) * srow[jn >> 7];       // GROUP_SIZE = 128
        }
        wrow[c] = (_Float16)val;
    }
}

// ---------------------------------------------------------------------------
// gfx1250 async global->LDS copy (ASYNCcnt-tracked), GVS addressing:
//   lds[VDST + ioffset] = mem[SADDR + VADDR + ioffset], 16 bytes per lane.
// ---------------------------------------------------------------------------
__device__ __forceinline__ void async_g2l_b128(unsigned lds_off, unsigned gbl_off,
                                               const void* sbase)
{
    asm volatile("global_load_async_to_lds_b128 %0, %1, %2"
                 :: "v"(lds_off), "v"(gbl_off), "s"(sbase)
                 : "memory");
}
__device__ __forceinline__ void wait_asynccnt0()
{
    asm volatile("s_wait_asynccnt 0x0" ::: "memory");
}
__device__ __forceinline__ unsigned lds_off_u32(const void* p)
{
    // Flat LDS aperture: addr[31:0] is the wave-relative LDS byte offset.
    return (unsigned)(unsigned long long)p;
}

// ---------------------------------------------------------------------------
// Pass 2 helpers: A tile (f32 -> f16) through registers.
// ---------------------------------------------------------------------------
__device__ __forceinline__ void load_regs_a(
    const float* __restrict__ X, int K, int mBase, int kOff, int tid,
    float4 (&ax)[4])
{
    const int arow = tid >> 3, ak4 = tid & 7;   // 32 rows x 8 float4-cols per pass
#pragma unroll
    for (int i = 0; i < 4; ++i)
        ax[i] = *(const float4*)(X + (size_t)(mBase + arow + 32 * i) * K + kOff + ak4 * 4);
}

__device__ __forceinline__ void store_lds_a(
    _Float16* __restrict__ la, int tid, const float4 (&ax)[4])
{
    const int arow = tid >> 3, ak4 = tid & 7;
#pragma unroll
    for (int i = 0; i < 4; ++i) {
        union { _Float16 h[4]; uint2 u; } cv;
        cv.h[0] = (_Float16)ax[i].x; cv.h[1] = (_Float16)ax[i].y;
        cv.h[2] = (_Float16)ax[i].z; cv.h[3] = (_Float16)ax[i].w;
        *(uint2*)(la + (arow + 32 * i) * LDSS + ak4 * 4) = cv.u;
    }
}

// B tile: issue 2 async 16-byte copies per thread (128 rows x 32 f16).
__device__ __forceinline__ void async_b_tile(
    const _Float16* __restrict__ Wh, int K, int nBase, int kOff, int tid,
    unsigned ldsb_off)
{
    const int brow = tid >> 2, bk4 = tid & 3;   // 64 rows x 4 b128-cols per pass
#pragma unroll
    for (int i = 0; i < 2; ++i) {
        const unsigned goff = (unsigned)(((size_t)(nBase + brow + 64 * i) * K
                                          + kOff + bk4 * 8) * sizeof(_Float16));
        const unsigned loff = ldsb_off
            + (unsigned)(((brow + 64 * i) * LDSS + bk4 * 8) * sizeof(_Float16));
        async_g2l_b128(loff, goff, (const void*)Wh);
    }
}

__global__ __launch_bounds__(256) void wmma_gemm_kernel(
    const float* __restrict__ X,        // (M, K) f32
    const _Float16* __restrict__ Wh,    // (N, K) f16
    const float* __restrict__ bias,     // (N)
    float* __restrict__ out,            // (M, N) f32
    int M, int N, int K)
{
    __shared__ __align__(16) _Float16 ldsA[2][BM * LDSS];
    __shared__ __align__(16) _Float16 ldsB[2][BN * LDSS];

    const int tid  = threadIdx.x;
    const int lane = tid & 31;
    const int wave = tid >> 5;
    const int h    = lane >> 4;     // lane half (ISA fragment layout)
    const int ln   = lane & 15;     // m (for A) / n (for B, C/D)
    const int waveM = wave & 3;     // 4 waves along M, 32 rows each
    const int waveN = wave >> 2;    // 2 waves along N, 64 cols each

    const int mBase = blockIdx.y * BM;
    const int nBase = blockIdx.x * BN;

    const unsigned ldsb_off[2] = { lds_off_u32(&ldsB[0][0]), lds_off_u32(&ldsB[1][0]) };

    v8f acc[2][4];
#pragma unroll
    for (int i = 0; i < 2; ++i)
#pragma unroll
        for (int j = 0; j < 4; ++j) acc[i][j] = {};

    const int nk = K / BK;
    float4 ax[4];

    // prologue: tile 0 -> LDS buffer 0
    async_b_tile(Wh, K, nBase, 0, tid, ldsb_off[0]);
    load_regs_a(X, K, mBase, 0, tid, ax);
    store_lds_a(ldsA[0], tid, ax);
    wait_asynccnt0();
    __syncthreads();

    for (int kt = 0; kt < nk; ++kt) {
        const int cur = kt & 1;
        if (kt + 1 < nk) {
            async_b_tile(Wh, K, nBase, (kt + 1) * BK, tid, ldsb_off[cur ^ 1]);
            load_regs_a(X, K, mBase, (kt + 1) * BK, tid, ax);
        }

        const _Float16* aB = &ldsA[cur][(waveM * 32) * LDSS];
        const _Float16* bB = &ldsB[cur][(waveN * 64) * LDSS];

        // A fragment (16x32 f16): lane (h,m) holds 16 contiguous K starting at 16h
        v16h afrag[2];
#pragma unroll
        for (int mi = 0; mi < 2; ++mi) {
            const _Float16* p = aB + (mi * 16 + ln) * LDSS;
            v8h lo = *(const v8h*)(p + h * 16);
            v8h hi = *(const v8h*)(p + h * 16 + 8);
            afrag[mi] = __builtin_shufflevector(lo, hi,
                0, 1, 2, 3, 4, 5, 6, 7, 8, 9, 10, 11, 12, 13, 14, 15);
        }
        // B fragment (32x16 f16): lane (h,n) holds 16 contiguous K starting at 16h,
        // B[k][n] = Wh[n][k] -> read row n of the (N,K)-layout LDS tile.
        v16h bfrag[4];
#pragma unroll
        for (int ni = 0; ni < 4; ++ni) {
            const _Float16* p = bB + (ni * 16 + ln) * LDSS + h * 16;
            bfrag[ni] = *(const v16h*)p;
        }

#pragma unroll
        for (int mi = 0; mi < 2; ++mi)
#pragma unroll
            for (int ni = 0; ni < 4; ++ni)
                acc[mi][ni] = __builtin_amdgcn_wmma_f32_16x16x32_f16(
                    false, afrag[mi], false, bfrag[ni],
                    (short)0, acc[mi][ni], false, false);

        if (kt + 1 < nk)
            store_lds_a(ldsA[cur ^ 1], tid, ax);

        wait_asynccnt0();   // own async-B chunks for the next buffer have landed
        __syncthreads();    // everyone's chunks (A stores + async B) visible
    }

    // epilogue: C/D layout -> element r lives at row (r + 8h), col ln
#pragma unroll
    for (int ni = 0; ni < 4; ++ni) {
        const int col = nBase + waveN * 64 + ni * 16 + ln;
        const float bv = bias[col];
#pragma unroll
        for (int mi = 0; mi < 2; ++mi) {
            const int rowBase = mBase + waveM * 32 + mi * 16 + 8 * h;
            float* p = out + (size_t)rowBase * N + col;
#pragma unroll
            for (int r = 0; r < 8; ++r) {
                *p = acc[mi][ni][r] + bv;
                p += N;
            }
        }
    }
}

// ---------------------------------------------------------------------------
extern "C" void kernel_launch(void* const* d_in, const int* in_sizes, int n_in,
                              void* d_out, int out_size, void* d_ws, size_t ws_size,
                              hipStream_t stream)
{
    const float* x       = (const float*)d_in[0];
    const int*   qweight = (const int*)  d_in[1];
    const float* scales  = (const float*)d_in[2];
    const int*   cidx    = (const int*)  d_in[3];
    const float* cw      = (const float*)d_in[4];
    const float* bias    = (const float*)d_in[5];
    float*       out     = (float*)d_out;

    // Derive shapes from flat sizes (matches setup_inputs()).
    const int O          = in_sizes[5];              // 4096
    const int nch        = in_sizes[4] / O;          // 40
    const int packed     = in_sizes[1] / O;          // 2028
    const int num_normal = packed * 2;               // 4056
    const int I          = num_normal + nch;         // 4096 (= K)
    const int n_groups   = in_sizes[2] / O;          // 32
    const int M          = in_sizes[0] / I;          // 8192 (= B*S)

    _Float16* Wh = (_Float16*)d_ws;  // needs O*I*2 = 33.5 MB of workspace

    dequant_rows_kernel<<<dim3(O), dim3(128), 0, stream>>>(
        qweight, scales, cidx, cw, Wh, O, I, packed, n_groups, nch);

    dim3 grid(O / BN, M / BM);   // exact multiples for these shapes
    wmma_gemm_kernel<<<grid, dim3(256), 0, stream>>>(x, Wh, bias, out, M, O, I);
}
